// SGCL_Encoder_73650099191970
// MI455X (gfx1250) — matrix operand order repaired and checked by reading, not compiled, not run on hardware
//
#include <hip/hip_runtime.h>

// ---------------------------------------------------------------------------
// SGCL encoder for MI455X (gfx1250, wave32, WMMA).
// GEMMs: v_wmma_f32_16x16x32_bf16 with bf16 hi/lo split (bf16x3) for ~fp32
// accuracy. LDS holds A/W tiles pre-swizzled into WMMA fragment order so each
// fragment load is two ds_load_b128. Loops are compile-time unrolled
// (template K/NC) so accumulators stay in VGPRs; the MMA loop iterates the
// output tile (nt) innermost so consecutive WMMAs hit independent
// accumulators (hides the 5-cycle WMMA->WMMA RAW hazard without v_nops).
// Edge softmax/aggregation use global atomics (max via order-preserving uint).
// ---------------------------------------------------------------------------

typedef __attribute__((ext_vector_type(16))) __bf16 v16bf;
typedef __attribute__((ext_vector_type(8)))  float  v8f;

union FragU { unsigned short u[16]; uint4 q[2]; v16bf v; };

__device__ inline unsigned short f2bf(float x) {
  unsigned u = __float_as_uint(x);
  u += 0x7FFFu + ((u >> 16) & 1u);           // round-to-nearest-even
  return (unsigned short)(u >> 16);
}
__device__ inline float bf2f(unsigned short h) {
  return __uint_as_float(((unsigned)h) << 16);
}

__device__ inline v8f wmma_bf16(v16bf a, v16bf b, v8f c) {
  // (neg_a, A, neg_b, B, c_mod, C, reuse_a, reuse_b)
  return __builtin_amdgcn_wmma_f32_16x16x32_bf16(false, a, false, b,
                                                 (short)0, c, false, false);
}

// Load one pre-swizzled fragment (16 bf16 = 32B contiguous, 16B aligned)
// -> two ds_load_b128 per lane.
__device__ inline v16bf ld_frag(const unsigned short* lds, int idx) {
  FragU f;
  const uint4* p = (const uint4*)(lds + idx);
  f.q[0] = p[0];
  f.q[1] = p[1];
  return f.v;
}

// ---------------------------------------------------------------------------
// bf16x3 GEMM:  C[nrows x NC] = act(A @ W + bias),  K in {128, 256}.
// A is given as one or two row-major [nrows x 128] blocks (A1 = second half
// of the K dimension, implementing the [h_pos | h_neg] concat for the MLP).
// Block: 256 threads = 8 waves; wave w owns rows [w*16, w*16+16) of a
// 128-row tile. LDS layout: frag-major, chunk*512 + lane*16 + j shorts.
// ---------------------------------------------------------------------------
template <int K, int NC, bool RELU>
__global__ __launch_bounds__(256) void gemm_bf16x3_kernel(
    const float* __restrict__ A0, const float* __restrict__ A1,
    const float* __restrict__ W, const float* __restrict__ bias,
    float* __restrict__ C, int nrows) {
  constexpr int NT  = NC / 16;        // 16-wide output tiles
  constexpr int NKC = K / 128;        // 128-deep K chunks
  constexpr int CSH = (NC == 128) ? 7 : 6;

  __shared__ unsigned short Ahi[8 * 4 * 512];   // [tile_m][ks][lane][16]
  __shared__ unsigned short Alo[8 * 4 * 512];
  __shared__ unsigned short Whi[NT * 4 * 512];  // [nt][ks][lane][16]
  __shared__ unsigned short Wlo[NT * 4 * 512];

  const int tid  = threadIdx.x;
  const int wave = tid >> 5;
  const int lane = tid & 31;
  const int row0 = blockIdx.x * 128;

  v8f zacc = {};
  v8f acc[NT];
#pragma unroll
  for (int i = 0; i < NT; ++i) acc[i] = zacc;

  for (int kc = 0; kc < NKC; ++kc) {
    const float* Asrc = (kc == 0) ? A0 : A1;   // both have row stride 128

    // ---- stage A chunk (128 rows x 128 k) into fragment order, hi/lo ----
    // pair of adjacent k's -> adjacent j's in the same lane -> one b32 store
    for (int p = tid; p < 128 * 64; p += 256) {
      int r  = p >> 6;
      int kk = (p & 63) << 1;
      int grow = row0 + r;
      float v0 = 0.f, v1 = 0.f;
      if (grow < nrows) {
        v0 = Asrc[grow * 128 + kk];
        v1 = Asrc[grow * 128 + kk + 1];
      }
      unsigned short h0 = f2bf(v0), h1 = f2bf(v1);
      unsigned short l0 = f2bf(v0 - bf2f(h0)), l1 = f2bf(v1 - bf2f(h1));
      int ks = kk >> 5, k32 = kk & 31;
      int hf = (k32 >> 3) & 1;
      int j  = (k32 & 7) | ((k32 >> 4) << 3);
      int idx = (((r >> 4) * 4 + ks) * 32 + ((r & 15) | (hf << 4))) * 16 + j;
      *(unsigned*)(&Ahi[idx]) = (unsigned)h0 | ((unsigned)h1 << 16);
      *(unsigned*)(&Alo[idx]) = (unsigned)l0 | ((unsigned)l1 << 16);
    }
    // ---- stage W chunk (128 k x NC) into fragment order, hi/lo ----
    // pair of adjacent k's in one column -> adjacent j's -> one b32 store
    for (int p = tid; p < 64 * NC; p += 256) {
      int c = p & (NC - 1);
      int k = (p >> CSH) << 1;
      float v0 = W[(kc * 128 + k) * NC + c];
      float v1 = W[(kc * 128 + k + 1) * NC + c];
      unsigned short h0 = f2bf(v0), h1 = f2bf(v1);
      unsigned short l0 = f2bf(v0 - bf2f(h0)), l1 = f2bf(v1 - bf2f(h1));
      int ks = k >> 5, k32 = k & 31;
      int hf = (k32 >> 3) & 1;
      int j  = (k32 & 7) | ((k32 >> 4) << 3);
      int idx = (((c >> 4) * 4 + ks) * 32 + ((c & 15) | (hf << 4))) * 16 + j;
      *(unsigned*)(&Whi[idx]) = (unsigned)h0 | ((unsigned)h1 << 16);
      *(unsigned*)(&Wlo[idx]) = (unsigned)l0 | ((unsigned)l1 << 16);
    }
    __syncthreads();

    // ---- A fragments for this wave's 16 rows (kept in VGPRs) ----
    v16bf a_hi[4], a_lo[4];
#pragma unroll
    for (int ks = 0; ks < 4; ++ks) {
      int idx = ((wave * 4 + ks) * 32 + lane) * 16;
      a_hi[ks] = ld_frag(Ahi, idx);
      a_lo[ks] = ld_frag(Alo, idx);
    }
    // ---- MMA: nt innermost so successive WMMAs target independent
    //      accumulators (acc[0..NT-1]); 3 bf16x3 terms per (ks,nt) ----
#pragma unroll
    for (int ks = 0; ks < 4; ++ks) {
#pragma unroll
      for (int nt = 0; nt < NT; ++nt) {
        int idx = ((nt * 4 + ks) * 32 + lane) * 16;
        v16bf bhi = ld_frag(Whi, idx);
        v16bf blo = ld_frag(Wlo, idx);
        acc[nt] = wmma_bf16(a_hi[ks], bhi, acc[nt]);   // hi*hi
        acc[nt] = wmma_bf16(a_hi[ks], blo, acc[nt]);   // hi*lo correction
        acc[nt] = wmma_bf16(a_lo[ks], bhi, acc[nt]);   // lo*hi correction
      }
    }
    __syncthreads();
  }

  // ---- write C: VGPR r of C/D holds M = r + 8*(lane>>4), N = lane&15 ----
  const int hl = lane >> 4, c0 = lane & 15;
#pragma unroll
  for (int nt = 0; nt < NT; ++nt) {
    int c = nt * 16 + c0;
    float bv = bias ? bias[c] : 0.f;
#pragma unroll
    for (int r = 0; r < 8; ++r) {
      int grow = row0 + wave * 16 + r + 8 * hl;
      if (grow < nrows) {
        float v = acc[nt][r] + bv;
        if (RELU) v = fmaxf(v, 0.f);
        C[grow * NC + c] = v;
      }
    }
  }
}

// ---------------------------------------------------------------------------
// Per-node attention logits: el[n,h] = sum_d feat[n,h,d]*al[h,d] (er likewise)
// ---------------------------------------------------------------------------
__global__ void eler_kernel(const float* __restrict__ feat,
                            const float* __restrict__ al,
                            const float* __restrict__ ar,
                            float* __restrict__ el, float* __restrict__ er,
                            int n) {
  int i = blockIdx.x * blockDim.x + threadIdx.x;
  if (i >= n * 8) return;
  int node = i >> 3, h = i & 7;
  const float* f  = feat + node * 128 + h * 16;
  const float* a0 = al + h * 16;
  const float* a1 = ar + h * 16;
  float sl = 0.f, sr = 0.f;
#pragma unroll
  for (int d = 0; d < 16; ++d) { sl += f[d] * a0[d]; sr += f[d] * a1[d]; }
  el[i] = sl;
  er[i] = sr;
}

// ---------------------------------------------------------------------------
// Init: segment-max keys -> 0 (== encoded -inf), segment-sums -> 0,
// and pre-seed the GAT outputs with their bias so aggregation is "+= only".
// ---------------------------------------------------------------------------
__global__ void init_kernel(unsigned* __restrict__ mP, float* __restrict__ sP,
                            unsigned* __restrict__ mN, float* __restrict__ sN,
                            float* __restrict__ hpos, float* __restrict__ hneg,
                            const float* __restrict__ bpos,
                            const float* __restrict__ bneg, int n) {
  int i = blockIdx.x * blockDim.x + threadIdx.x;
  if (i >= n * 128) return;
  int node = i >> 7, c = i & 127;
  hpos[i] = bpos[c];
  hneg[i] = bneg[c];
  if (c < 8) {
    mP[node * 8 + c] = 0u;  sP[node * 8 + c] = 0.f;
    mN[node * 8 + c] = 0u;  sN[node * 8 + c] = 0.f;
  }
}

// order-preserving float<->uint (for atomicMax over signed floats)
__device__ inline unsigned fenc(float f) {
  unsigned u = __float_as_uint(f);
  return (u >> 31) ? ~u : (u | 0x80000000u);
}
__device__ inline float fdec(unsigned k) {
  unsigned u = (k & 0x80000000u) ? (k ^ 0x80000000u) : ~k;
  return __uint_as_float(u);
}

__device__ inline float edge_logit(const float* el, const float* er, int s,
                                   int d, int h) {
  float x = el[s * 8 + h] + er[d * 8 + h];
  return x > 0.f ? x : 0.2f * x;  // leaky_relu(0.2)
}

__global__ void edge_max_kernel(const int* __restrict__ src,
                                const int* __restrict__ dst,
                                const float* __restrict__ el,
                                const float* __restrict__ er,
                                unsigned* __restrict__ m, int eh) {
  int i = blockIdx.x * blockDim.x + threadIdx.x;
  if (i >= eh) return;
  int e = i >> 3, h = i & 7;
  int s = src[e], d = dst[e];
  atomicMax(&m[d * 8 + h], fenc(edge_logit(el, er, s, d, h)));
}

__global__ void edge_sum_kernel(const int* __restrict__ src,
                                const int* __restrict__ dst,
                                const float* __restrict__ el,
                                const float* __restrict__ er,
                                const unsigned* __restrict__ m,
                                float* __restrict__ ssum, int eh) {
  int i = blockIdx.x * blockDim.x + threadIdx.x;
  if (i >= eh) return;
  int e = i >> 3, h = i & 7;
  int s = src[e], d = dst[e];
  float x = edge_logit(el, er, s, d, h);
  atomicAdd(&ssum[d * 8 + h], __expf(x - fdec(m[d * 8 + h])));
}

__global__ void edge_agg_kernel(const int* __restrict__ src,
                                const int* __restrict__ dst,
                                const float* __restrict__ el,
                                const float* __restrict__ er,
                                const unsigned* __restrict__ m,
                                const float* __restrict__ ssum,
                                const float* __restrict__ feat,
                                float* __restrict__ hout, int eh) {
  int i = blockIdx.x * blockDim.x + threadIdx.x;
  if (i >= eh) return;
  int e = i >> 3, h = i & 7;
  int s = src[e], d = dst[e];
  float x = edge_logit(el, er, s, d, h);
  float w = __expf(x - fdec(m[d * 8 + h])) / ssum[d * 8 + h];
  const float* fs = feat + s * 128 + h * 16;
  float* o = hout + d * 128 + h * 16;
#pragma unroll
  for (int t = 0; t < 16; ++t) atomicAdd(&o[t], w * fs[t]);
}

// ---------------------------------------------------------------------------
extern "C" void kernel_launch(void* const* d_in, const int* in_sizes, int n_in,
                              void* d_out, int out_size, void* d_ws,
                              size_t ws_size, hipStream_t stream) {
  const float* features = (const float*)d_in[0];
  const int*   pos_src  = (const int*)d_in[1];
  const int*   pos_dst  = (const int*)d_in[2];
  const int*   neg_src  = (const int*)d_in[3];
  const int*   neg_dst  = (const int*)d_in[4];
  const float* W_pos = (const float*)d_in[5];
  const float* al_pos = (const float*)d_in[6];
  const float* ar_pos = (const float*)d_in[7];
  const float* b_pos = (const float*)d_in[8];
  const float* W_neg = (const float*)d_in[9];
  const float* al_neg = (const float*)d_in[10];
  const float* ar_neg = (const float*)d_in[11];
  const float* b_neg = (const float*)d_in[12];
  const float* W1 = (const float*)d_in[13];
  const float* b1 = (const float*)d_in[14];
  const float* W2 = (const float*)d_in[15];
  const float* b2 = (const float*)d_in[16];

  const int N = in_sizes[0] / 128;
  const int E = in_sizes[1];

  // workspace layout (floats)
  float* ws    = (float*)d_ws;
  float* featP = ws;                 // N*128
  float* featN = featP + (size_t)N * 128;
  float* elP   = featN + (size_t)N * 128;  // N*8 each below
  float* erP   = elP + (size_t)N * 8;
  float* elN   = erP + (size_t)N * 8;
  float* erN   = elN + (size_t)N * 8;
  unsigned* mP = (unsigned*)(erN + (size_t)N * 8);
  float*    sP = (float*)(mP + (size_t)N * 8);
  unsigned* mN = (unsigned*)(sP + (size_t)N * 8);
  float*    sN = (float*)(mN + (size_t)N * 8);
  float* tmid  = featP;              // reuse featP after aggregation for MLP

  float* hpos = (float*)d_out;
  float* hneg = hpos + (size_t)N * 128;
  float* hfin = hneg + (size_t)N * 128;

  const int gemm_grid = (N + 127) / 128;
  const int eh        = E * 8;
  const int edge_grid = (eh + 255) / 256;

  // 1) feature transforms (bf16x3 WMMA GEMMs)
  gemm_bf16x3_kernel<128, 128, false><<<gemm_grid, 256, 0, stream>>>(
      features, nullptr, W_pos, nullptr, featP, N);
  gemm_bf16x3_kernel<128, 128, false><<<gemm_grid, 256, 0, stream>>>(
      features, nullptr, W_neg, nullptr, featN, N);

  // 2) per-node attention logits
  eler_kernel<<<(N * 8 + 255) / 256, 256, 0, stream>>>(featP, al_pos, ar_pos,
                                                       elP, erP, N);
  eler_kernel<<<(N * 8 + 255) / 256, 256, 0, stream>>>(featN, al_neg, ar_neg,
                                                       elN, erN, N);

  // 3) init stats + seed outputs with bias
  init_kernel<<<(N * 128 + 255) / 256, 256, 0, stream>>>(
      mP, sP, mN, sN, hpos, hneg, b_pos, b_neg, N);

  // 4) edge softmax (max, sum) + weighted aggregation, per sign
  edge_max_kernel<<<edge_grid, 256, 0, stream>>>(pos_src, pos_dst, elP, erP, mP, eh);
  edge_max_kernel<<<edge_grid, 256, 0, stream>>>(neg_src, neg_dst, elN, erN, mN, eh);
  edge_sum_kernel<<<edge_grid, 256, 0, stream>>>(pos_src, pos_dst, elP, erP, mP, sP, eh);
  edge_sum_kernel<<<edge_grid, 256, 0, stream>>>(neg_src, neg_dst, elN, erN, mN, sN, eh);
  edge_agg_kernel<<<edge_grid, 256, 0, stream>>>(pos_src, pos_dst, elP, erP, mP, sP, featP, hpos, eh);
  edge_agg_kernel<<<edge_grid, 256, 0, stream>>>(neg_src, neg_dst, elN, erN, mN, sN, featN, hneg, eh);

  // 5) MLP: relu([h_pos|h_neg] @ W1 + b1) @ W2 + b2
  gemm_bf16x3_kernel<256, 128, true><<<gemm_grid, 256, 0, stream>>>(
      hpos, hneg, W1, b1, tmid, N);
  gemm_bf16x3_kernel<128, 64, false><<<gemm_grid, 256, 0, stream>>>(
      tmid, nullptr, W2, b2, hfin, N);
}
